// EchoStateNetwork_24215025615216
// MI455X (gfx1250) — matrix operand order repaired and checked by reading, not compiled, not run on hardware
//
#include <hip/hip_runtime.h>
#include <math.h>

// ---------------------------------------------------------------------------
// Echo State Network on MI455X (gfx1250, wave32, WMMA).
//
// Key trick: state-trace slot layout mirrors the weight layout.
//   Wcat[n][k]  = bf16 [ W_in row n (256) | W row n (2048) ]     (k = 0..2303)
//   slot t row b = bf16 [ u_t[b] (256)    | s_t[b] (2048) ]      (k = 0..2303)
// so the step GEMM  pre = [u_t|s_t] @ Wcat^T  is ONE uniform K=2304 loop:
// constant trip counts, identical for all split-K waves -> fully pipelined
// load/WMMA codegen (no exec-mask loops).
//
//   phase 0:  build Wcat/wroB (bf16), scatter u into the u-part of every slot,
//             write initial state into slot 0's s-part (+ fp32 copy).
//   phase 1 (x512): s_new = 0.7*s + 0.3*tanh([u_t|s_t] @ Wcat^T + bias)
//             4 waves split K per 16x16 tile (18 WMMAs each), LDS reduction,
//             cooperative tanh/leak epilogue; 1024 waves / 128 blocks per step.
//   phase 2:  out = states @ w_ro^T + b_ro  (16384x2048x256), 2 n-tiles/wave.
//
// Weights + bf16 state trace (~86 MB) stay resident in the 192 MB L2; the
// serial 512-step scan is the latency floor, so the step kernel minimizes
// per-step dependency-chain depth.
// ---------------------------------------------------------------------------

typedef __attribute__((ext_vector_type(16))) __bf16 v16bf;
typedef __attribute__((ext_vector_type(8)))  __bf16 v8bf;
typedef __attribute__((ext_vector_type(8)))  float  v8f;

#define INPUT_DIM 256
#define RES_DIM   2048
#define OUT_DIM   256
#define BATCH     32
#define SEQ       512
#define KCAT      (INPUT_DIM + RES_DIM)   // 2304 (row length of Wcat and slots)
#define KSPLIT    (KCAT / 4)              // 576 K per wave in the step kernel
#define LEAK      0.3f
#define SLOTC     (BATCH * KCAT)          // 73728 elems per [u|s] slot

// A-fragment (16x32 bf16, row-major source):
//   lanes 0-15  (lh=0): M=lane,    K = {k0+0..15}
//   lanes 16-31 (lh=1): M=lane-16, K = {k0+16..31}
static __device__ __forceinline__ v16bf load_a_frag(const __bf16* rowPtrAtK0, int lh) {
  const v8bf c0 = *(const v8bf*)(rowPtrAtK0 + lh * 16);
  const v8bf c1 = *(const v8bf*)(rowPtrAtK0 + lh * 16 + 8);
  return __builtin_shufflevector(c0, c1, 0,1,2,3,4,5,6,7,8,9,10,11,12,13,14,15);
}
// B-fragment (32x16 bf16): lane = column n = lane&15, 16 contiguous K starting
// at k0 + (lane>=16 ? 16 : 0); source is n-major [n][k] -> one 32-byte load.

// --------------------------- prep kernels ----------------------------------

__global__ void prep_wcat_kernel(const float* __restrict__ Win,
                                 const float* __restrict__ W,
                                 __bf16* __restrict__ Wcat) {
  const int idx = blockIdx.x * blockDim.x + threadIdx.x;     // n*KCAT + k
  if (idx >= RES_DIM * KCAT) return;
  const int n = idx / KCAT;
  const int k = idx - n * KCAT;
  const float v = (k < INPUT_DIM) ? Win[(size_t)n * INPUT_DIM + k]
                                  : W[(size_t)n * RES_DIM + (k - INPUT_DIM)];
  Wcat[idx] = (__bf16)v;
}

__global__ void conv_bf16_kernel(const float* __restrict__ src,
                                 __bf16* __restrict__ dst, int n) {
  const int idx = blockIdx.x * blockDim.x + threadIdx.x;
  if (idx < n) dst[idx] = (__bf16)src[idx];
}

// Scatter u[b][t][i] (fp32) into the u-part of slot t: slots[t][b][i], i<256.
__global__ void pack_u_kernel(const float* __restrict__ u,
                              __bf16* __restrict__ slots) {
  const int idx = blockIdx.x * blockDim.x + threadIdx.x;
  if (idx >= SEQ * BATCH * INPUT_DIM) return;
  const int t   = idx / (BATCH * INPUT_DIM);
  const int rem = idx - t * (BATCH * INPUT_DIM);
  const int b   = rem / INPUT_DIM;
  const int i   = rem - b * INPUT_DIM;
  slots[(size_t)t * SLOTC + (size_t)b * KCAT + i] =
      (__bf16)u[((size_t)b * SEQ + t) * INPUT_DIM + i];
}

// Initial state into slot 0's s-part + fp32 ping buffer.
__global__ void init_state_kernel(const float* __restrict__ s0,
                                  __bf16* __restrict__ slots,
                                  float* __restrict__ sF32) {
  const int idx = blockIdx.x * blockDim.x + threadIdx.x;
  if (idx >= BATCH * RES_DIM) return;
  const int b = idx / RES_DIM;
  const int r = idx - b * RES_DIM;
  const float v = s0[idx];
  slots[(size_t)b * KCAT + INPUT_DIM + r] = (__bf16)v;
  sF32[idx] = v;
}

// --------------------------- recurrent step --------------------------------
// 256 output tiles (2 m x 128 n), 4 waves split K per tile -> 1024 waves.
// grid: 128 blocks x 256 threads (8 waves = 2 tiles per block).

__global__ __launch_bounds__(256)
void esn_step_kernel(const __bf16* __restrict__ Wcat,    // [2048][2304]
                     const __bf16* __restrict__ slots,   // [T+1][32][2304]
                     const float*  __restrict__ bias,    // [2048]
                     const float*  __restrict__ sOld,    // [32][2048] fp32
                     float*        __restrict__ sNew,    // [32][2048] fp32
                     __bf16*       __restrict__ slotNext,// slots + (t+1)*SLOTC
                     int t) {
  __shared__ float partial[2][4][256];   // [tileInBlock][kSplitWave][elem]

  const int tid   = threadIdx.x;
  const int lane  = tid & 31;
  const int lm    = lane & 15;
  const int lh    = lane >> 4;
  const int wib   = tid >> 5;            // wave in block: 0..7
  const int tileL = wib >> 2;            // tile within block: 0..1
  const int wK    = wib & 3;             // K-split slice: 0..3
  const int tile  = blockIdx.x * 2 + tileL;     // 0..255
  const int mbase = (tile & 1) * 16;            // batch tile
  const int nbase = (tile >> 1) * 16;           // reservoir tile

  const int kStart = wK * KSPLIT;
  const __bf16* aR = slots + (size_t)t * SLOTC
                           + (size_t)(mbase + lm) * KCAT + kStart;
  const __bf16* bW = Wcat + (size_t)(nbase + lm) * KCAT + lh * 16 + kStart;

  // One uniform K slice: constant 18 iterations for every wave.
  v8f acc = {};
  #pragma unroll
  for (int k0 = 0; k0 < KSPLIT; k0 += 32) {
    v16bf a = load_a_frag(aR + k0, lh);
    v16bf b = *(const v16bf*)(bW + k0);
    acc = __builtin_amdgcn_wmma_f32_16x16x32_bf16(false, a, false, b,
                                                  (short)0, acc, false, false);
  }

  // Stash this wave's partial: C/D layout -> elem e = (g + lh*8)*16 + (lane&15)
  #pragma unroll
  for (int g = 0; g < 8; ++g) {
    partial[tileL][wK][(g + lh * 8) * 16 + lm] = acc[g];
  }
  __syncthreads();

  // Cross-wave reduction + leak/tanh epilogue: 128 threads per tile, 2 elems each.
  const int eBase = (tid & 127) * 2;
  #pragma unroll
  for (int j = 0; j < 2; ++j) {
    const int e = eBase + j;
    const float pre = partial[tileL][0][e] + partial[tileL][1][e]
                    + partial[tileL][2][e] + partial[tileL][3][e]
                    + bias[nbase + (e & 15)];
    const int b = mbase + (e >> 4);
    const int r = nbase + (e & 15);
    const float so = sOld[(size_t)b * RES_DIM + r];
    const float sn = (1.0f - LEAK) * so + LEAK * tanhf(pre);
    sNew[(size_t)b * RES_DIM + r] = sn;
    slotNext[(size_t)b * KCAT + INPUT_DIM + r] = (__bf16)sn;
  }
}

// ------------------------------ readout ------------------------------------
// GEMM: rows p = t*32+b (16384), cols o (256), K = 2048 (s-part of slots).
// Each wave: 1 m-tile x 2 n-tiles (A-fragment reused, 2 independent accs).
// grid: 1024 blocks x 256 threads = 8192 waves = 1024 m-tiles x 8 n-pairs.

__global__ __launch_bounds__(256)
void esn_readout_kernel(const __bf16* __restrict__ slots,  // [T+1][32][2304]
                        const __bf16* __restrict__ wroB,   // [256][2048]
                        const float*  __restrict__ bro,    // [256]
                        float* __restrict__ out) {         // [32][512][256]
  const int lane  = threadIdx.x & 31;
  const int lm    = lane & 15;
  const int lh    = lane >> 4;
  const int wave  = (blockIdx.x * blockDim.x + threadIdx.x) >> 5;
  const int n0    = (wave & 7) * 32;    // pair of n-tiles: n0, n0+16
  const int pbase = (wave >> 3) * 16;   // row tile; stays inside one t (B=32)
  const int tt    = pbase >> 5;
  const int b0    = pbase & 31;

  const __bf16* aS  = slots + (size_t)(tt + 1) * SLOTC
                            + (size_t)(b0 + lm) * KCAT + INPUT_DIM;
  const __bf16* bW0 = wroB + (size_t)(n0 + lm) * RES_DIM + lh * 16;
  const __bf16* bW1 = bW0 + (size_t)16 * RES_DIM;

  v8f acc0 = {};
  v8f acc1 = {};
  #pragma unroll 4
  for (int k0 = 0; k0 < RES_DIM; k0 += 32) {
    v16bf a  = load_a_frag(aS + k0, lh);
    v16bf b0 = *(const v16bf*)(bW0 + k0);
    v16bf b1 = *(const v16bf*)(bW1 + k0);
    acc0 = __builtin_amdgcn_wmma_f32_16x16x32_bf16(false, a, false, b0,
                                                   (short)0, acc0, false, false);
    acc1 = __builtin_amdgcn_wmma_f32_16x16x32_bf16(false, a, false, b1,
                                                   (short)0, acc1, false, false);
  }

  #pragma unroll
  for (int g = 0; g < 8; ++g) {
    const int p = pbase + g + lh * 8;
    const int t = p >> 5;
    const int b = p & 31;
    const size_t rowOff = ((size_t)b * SEQ + t) * OUT_DIM;
    const int o0 = n0 + lm;
    const int o1 = o0 + 16;
    out[rowOff + o0] = acc0[g] + bro[o0];
    out[rowOff + o1] = acc1[g] + bro[o1];
  }
}

// ------------------------------ launcher -----------------------------------

extern "C" void kernel_launch(void* const* d_in, const int* in_sizes, int n_in,
                              void* d_out, int out_size, void* d_ws, size_t ws_size,
                              hipStream_t stream) {
  const float* u    = (const float*)d_in[0];  // [32][512][256]
  const float* st0  = (const float*)d_in[1];  // [32][2048]
  const float* Win  = (const float*)d_in[2];  // [2048][256]
  const float* W    = (const float*)d_in[3];  // [2048][2048]
  const float* bias = (const float*)d_in[4];  // [2048]
  const float* wro  = (const float*)d_in[5];  // [256][2048]
  const float* bro  = (const float*)d_in[6];  // [256]
  float* out = (float*)d_out;

  // Workspace layout (256B-aligned offsets), total ~86.7 MB:
  char* ws = (char*)d_ws;
  __bf16* Wcat  = (__bf16*)(ws);                        //  9,437,184 B
  __bf16* wroB  = (__bf16*)(ws +  9437184);             //  1,048,576 B
  __bf16* slots = (__bf16*)(ws + 10485760);             // 75,644,928 B (513 slots)
  float*  sF32  = (float*) (ws + 86130688);             //  524,288 B (ping/pong)

  // phase 0: weight conversion + u scatter + state init
  prep_wcat_kernel<<<(RES_DIM * KCAT + 255) / 256, 256, 0, stream>>>(Win, W, Wcat);
  conv_bf16_kernel<<<(OUT_DIM * RES_DIM + 255) / 256, 256, 0, stream>>>(
      wro, wroB, OUT_DIM * RES_DIM);
  pack_u_kernel<<<(SEQ * BATCH * INPUT_DIM + 255) / 256, 256, 0, stream>>>(u, slots);
  init_state_kernel<<<(BATCH * RES_DIM + 255) / 256, 256, 0, stream>>>(
      st0, slots, sF32);

  // phase 1: 512 sequential recurrent steps (graph-replay amortizes launches)
  for (int t = 0; t < SEQ; ++t) {
    const float* so = sF32 + (size_t)(t & 1) * (BATCH * RES_DIM);
    float*       sn = sF32 + (size_t)((t + 1) & 1) * (BATCH * RES_DIM);
    __bf16* slotNxt = slots + (size_t)(t + 1) * SLOTC;
    esn_step_kernel<<<128, 256, 0, stream>>>(Wcat, slots + (size_t)t * SLOTC - 0,
                                             bias, so, sn, slotNxt, 0);
  }

  // phase 2: batched readout GEMM over the whole state trace
  esn_readout_kernel<<<1024, 256, 0, stream>>>(slots, wroB, bro, out);
}
